// PointNet_32744830665464
// MI455X (gfx1250) — compile-verified, matrix operand order checked by reading, not run on hardware
//
#include <hip/hip_runtime.h>
#include <hip/hip_bf16.h>

typedef __attribute__((ext_vector_type(16))) _Float16 v16h;
typedef __attribute__((ext_vector_type(8)))  float    v8f;

__device__ __forceinline__ v8f wmma_f16(v16h a, v16h b, v8f c) {
  // D = A(16x32 f16) x B(32x16 f16) + C(16x16 f32)
  return __builtin_amdgcn_wmma_f32_16x16x32_f16(false, a, false, b, (short)0, c, false, false);
}

__device__ __forceinline__ v16h vzero16h() {
  v16h z;
#pragma unroll
  for (int i = 0; i < 16; ++i) z[i] = (_Float16)0.0f;
  return z;
}

// Gather one B fragment (KxN = 32x16, f16) from an LDS weight tile stored
// row-major [K][32]. ISA 16-bit layout: lane l holds column n=(l&15)+16*tile,
// halfs 0..7 -> K = kbase+8*hi .. +7, halfs 8..15 -> K = kbase+16+8*hi .. +7.
__device__ __forceinline__ v16h load_b_frag(const _Float16* W, int kbase, int tile, int lane) {
  const int n  = (lane & 15) + 16 * tile;
  const int hi = (lane >> 4) & 1;
  v16h b;
#pragma unroll
  for (int i = 0; i < 16; ++i) {
    const int k = kbase + ((i < 8) ? (8 * hi + i) : (16 + 8 * hi + (i - 8)));
    b[i] = W[k * 32 + n];
  }
  return b;
}

// One PointNet message-passing layer:
//   msg = relu([feat | pos_s - pos_d] @ Wa + ba) @ Wb + bb    (per edge)
//   agg[d] = max(agg[d], msg)  via int-bit atomic max (agg pre-zeroed == post-relu)
// CIN = 6 (pos features, layer 1) or 35 (hidden f16 32 + pos diff 3, layer 2).
template <int CIN, int KPAD>
__global__ __launch_bounds__(128) void edge_mlp_scatter_max(
    const float* __restrict__ pos, const int* __restrict__ src, const int* __restrict__ dst,
    const _Float16* __restrict__ hinh,   // f16 mirror of previous layer (layer 2 only)
    const float* __restrict__ Wa, const float* __restrict__ ba,
    const float* __restrict__ Wb, const float* __restrict__ bb,
    int* __restrict__ agg, int nEdges) {
  __shared__ _Float16 sWa[KPAD * 32];
  __shared__ _Float16 sWb[32 * 32];
  __shared__ float sBa[32], sBb[32];
  __shared__ __align__(16) _Float16 sX[4][16 * 32];  // per-wave hidden tile
  __shared__ __align__(16) int sDst[4][16];

  const int tid = threadIdx.x;
  // Stage padded f16 weights in LDS (rows >= CIN are zero).
  for (int i = tid; i < KPAD * 32; i += blockDim.x) {
    const int k = i >> 5, n = i & 31;
    sWa[i] = (_Float16)((k < CIN) ? Wa[k * 32 + n] : 0.0f);
  }
  for (int i = tid; i < 32 * 32; i += blockDim.x) sWb[i] = (_Float16)Wb[i];
  if (tid < 32) { sBa[tid] = ba[tid]; sBb[tid] = bb[tid]; }
  __syncthreads();

  const int lane = tid & 31;
  const int wv   = tid >> 5;
  const int m    = lane & 15;
  const int hi   = lane >> 4;

  // Loop-invariant B fragments live in registers for the whole kernel.
  v16h Ba[KPAD / 32][2];
#pragma unroll
  for (int kc = 0; kc < KPAD / 32; ++kc)
#pragma unroll
    for (int t = 0; t < 2; ++t) Ba[kc][t] = load_b_frag(sWa, kc * 32, t, lane);
  v16h Bb[2];
#pragma unroll
  for (int t = 0; t < 2; ++t) Bb[t] = load_b_frag(sWb, 0, t, lane);

  // Hoist biases (per-lane column m / m+16) into registers: no LDS traffic in loop.
  const float biasA0 = sBa[m], biasA1 = sBa[m + 16];
  const float biasB0 = sBb[m], biasB1 = sBb[m + 16];

  const int nGroups = (nEdges + 15) >> 4;
  const int wstride = (int)gridDim.x * (blockDim.x >> 5);
  _Float16* X = sX[wv];
  const uint4* Xv = (const uint4*)X;          // 4 x uint4 per 32-half row
  const int xi0 = m * 4 + hi;                 // K 8*hi .. 8*hi+7
  const int xi1 = m * 4 + 2 + hi;             // K 16+8*hi .. +7
  const int4* dsv = (const int4*)&sDst[wv][0];
  const int KC1 = (KPAD / 32) - 1;            // second k-chunk index when KPAD==64

  for (int grp = blockIdx.x * (blockDim.x >> 5) + wv; grp < nGroups; grp += wstride) {
    const int base  = grp << 4;
    const int e     = base + m;
    const bool valid = e < nEdges;

    int s = 0;
    float sx=0.f, sy=0.f, sz=0.f, dx=0.f, dy=0.f, dz=0.f;
    if (valid && (hi == 0 || CIN > 8)) s = src[e];
    if (valid && hi == 0) {
      const int d = dst[e];
      sDst[wv][m] = d;
      sx = pos[3*s]; sy = pos[3*s+1]; sz = pos[3*s+2];
      dx = sx - pos[3*d]; dy = sy - pos[3*d+1]; dz = sz - pos[3*d+2];
    }

    // ---- A fragment, k-chunk 0 (K 0..31) ----
    v16h a0;
    if (CIN <= 8) {
      a0 = vzero16h();
      if (valid && hi == 0) {  // K=0..5 live in halfs 0..5 of the hi==0 lane
        a0[0]=(_Float16)sx; a0[1]=(_Float16)sy; a0[2]=(_Float16)sz;
        a0[3]=(_Float16)dx; a0[4]=(_Float16)dy; a0[5]=(_Float16)dz;
      }
    } else {
      // Hidden features from f16 mirror: two 16B loads, zero conversions.
      union { uint4 u[2]; v16h v; } au;
      if (valid) {
        const uint4* hr = (const uint4*)(hinh + (size_t)s * 32);  // 4 x uint4 per node
        au.u[0] = hr[hi];        // halfs 8*hi .. 8*hi+7
        au.u[1] = hr[2 + hi];    // halfs 16+8*hi .. +7
        a0 = au.v;
      } else {
        a0 = vzero16h();
      }
    }

    // ---- GEMM1: hidden = relu(A @ Wa + ba) ----
    v8f c0 = {}; v8f c1 = {};
    c0 = wmma_f16(a0, Ba[0][0], c0);
    c1 = wmma_f16(a0, Ba[0][1], c1);
    if (KPAD == 64) {  // k-chunk 1: features 32..34 = pos diff
      v16h a1 = vzero16h();
      if (valid && hi == 0) { a1[0]=(_Float16)dx; a1[1]=(_Float16)dy; a1[2]=(_Float16)dz; }
      c0 = wmma_f16(a1, Ba[KC1][0], c0);
      c1 = wmma_f16(a1, Ba[KC1][1], c1);
    }

    // Bias + relu; deposit hidden tile (16 edges x 32) into per-wave LDS.
    // C/D layout: lane holds column n=m (tile0) / m+16 (tile1), rows r+8*hi.
#pragma unroll
    for (int r = 0; r < 8; ++r) {
      float v0 = c0[r] + biasA0; v0 = v0 > 0.f ? v0 : 0.f;
      float v1 = c1[r] + biasA1; v1 = v1 > 0.f ? v1 : 0.f;
      const int row = r + 8 * hi;
      X[row * 32 + m]      = (_Float16)v0;
      X[row * 32 + m + 16] = (_Float16)v1;
    }
    asm volatile("" ::: "memory");  // LDS ops are in-order within a wave

    // Re-load hidden tile in A layout: two aligned 16B LDS reads per lane.
    union { uint4 u[2]; v16h h; } a2u;
    a2u.u[0] = Xv[xi0];
    a2u.u[1] = Xv[xi1];
    const v16h a2 = a2u.h;

    // ---- GEMM2: msg = hidden @ Wb + bb ----
    v8f d0 = {}; v8f d1 = {};
    d0 = wmma_f16(a2, Bb[0], d0);
    d1 = wmma_f16(a2, Bb[1], d1);

    // Vectorized read of the 8 destination ids this lane scatters to.
    int dvals[8];
    *(int4*)&dvals[0] = dsv[2 * hi];
    *(int4*)&dvals[4] = dsv[2 * hi + 1];

    // Scatter-max (relu folded: agg pre-zeroed, int-bit max is exact since
    // running max starts at +0.0 and int ordering matches on non-negatives).
#pragma unroll
    for (int r = 0; r < 8; ++r) {
      const int e2 = base + r + 8 * hi;
      if (e2 < nEdges) {
        const int dn = dvals[r];
        atomicMax(&agg[dn * 32 + m],      __float_as_int(d0[r] + biasB0));
        atomicMax(&agg[dn * 32 + m + 16], __float_as_int(d1[r] + biasB1));
      }
    }
  }
}

// Convert fp32 aggregation buffer to a packed f16 mirror (4 values / thread).
__global__ void cvt_f32_to_f16(const float4* __restrict__ in, uint2* __restrict__ outp, int n4) {
  const int i = blockIdx.x * blockDim.x + threadIdx.x;
  if (i < n4) {
    const float4 v = in[i];
    union { _Float16 h[4]; uint2 u; } p;
    p.h[0] = (_Float16)v.x; p.h[1] = (_Float16)v.y;
    p.h[2] = (_Float16)v.z; p.h[3] = (_Float16)v.w;
    outp[i] = p.u;
  }
}

// Global max pool per graph: h2 >= 0 and g pre-zeroed, so int-bit max is exact.
__global__ void pool_kernel(const int* __restrict__ h2bits, const int* __restrict__ batch,
                            int* __restrict__ g, int nNodes) {
  const int i = blockIdx.x * blockDim.x + threadIdx.x;
  if (i < nNodes * 32) {
    const int node = i >> 5, c = i & 31;
    atomicMax(&g[batch[node] * 32 + c], h2bits[i]);
  }
}

// out[b] = g[b,:] @ Wout + bout  (fp32, tiny)
__global__ void out_kernel(const int* __restrict__ gbits, const float* __restrict__ Wout,
                           const float* __restrict__ bout, float* __restrict__ out, int nGraphs) {
  const int b = blockIdx.x * blockDim.x + threadIdx.x;
  if (b < nGraphs) {
    float acc = bout[0];
#pragma unroll
    for (int c = 0; c < 32; ++c) acc += __int_as_float(gbits[b * 32 + c]) * Wout[c];
    out[b] = acc;
  }
}

extern "C" void kernel_launch(void* const* d_in, const int* in_sizes, int n_in,
                              void* d_out, int out_size, void* d_ws, size_t ws_size,
                              hipStream_t stream) {
  const float* pos   = (const float*)d_in[0];
  const int*   eidx  = (const int*)d_in[1];
  const int*   batch = (const int*)d_in[2];
  const float* W1a = (const float*)d_in[3];  const float* b1a = (const float*)d_in[4];
  const float* W1b = (const float*)d_in[5];  const float* b1b = (const float*)d_in[6];
  const float* W2a = (const float*)d_in[7];  const float* b2a = (const float*)d_in[8];
  const float* W2b = (const float*)d_in[9];  const float* b2b = (const float*)d_in[10];
  const float* Wout = (const float*)d_in[11]; const float* bout = (const float*)d_in[12];

  const int N = in_sizes[0] / 3;   // nodes
  const int E = in_sizes[1] / 2;   // edges
  const int nGraphs = out_size;    // [B,1] output

  const int* src = eidx;
  const int* dst = eidx + E;

  // Workspace layout (floats stored as int bit patterns for atomic max):
  int*      h1  = (int*)d_ws;                       // [N][32] fp32 agg, layer 1
  int*      h2  = h1 + (size_t)N * 32;              // [N][32] fp32 agg, layer 2
  int*      g   = h2 + (size_t)N * 32;              // [B][32] pooled
  _Float16* h1h = (_Float16*)(g + (size_t)nGraphs * 32);  // [N][32] f16 mirror of h1
  const size_t zbytes = ((size_t)2 * N * 32 + (size_t)nGraphs * 32) * sizeof(int);
  hipMemsetAsync(d_ws, 0, zbytes, stream);

  const int threads = 128;         // 4 waves per block
  const int blocks  = 4096;        // grid-stride over E/16 edge groups

  edge_mlp_scatter_max<6, 32><<<blocks, threads, 0, stream>>>(
      pos, src, dst, nullptr, W1a, b1a, W1b, b1b, h1, E);

  const int n4 = (N * 32) / 4;
  cvt_f32_to_f16<<<(n4 + 255) / 256, 256, 0, stream>>>(
      (const float4*)h1, (uint2*)h1h, n4);

  edge_mlp_scatter_max<35, 64><<<blocks, threads, 0, stream>>>(
      pos, src, dst, h1h, W2a, b2a, W2b, b2b, h2, E);

  const int ptotal = N * 32;
  pool_kernel<<<(ptotal + 255) / 256, 256, 0, stream>>>(h2, batch, g, N);
  out_kernel<<<1, 256, 0, stream>>>(g, Wout, bout, (float*)d_out, nGraphs);
}
